// HarModel_23854248362135
// MI455X (gfx1250) — compile-verified, matrix operand order checked by reading, not run on hardware
//
#include <hip/hip_runtime.h>
#include <hip/hip_bf16.h>
#include <stdint.h>

// ---------------------------------------------------------------------------
// Types for CDNA5 WMMA (wave32, 16x16x32 bf16 -> f32 accumulate)
// ---------------------------------------------------------------------------
typedef __bf16 v16bf __attribute__((ext_vector_type(16)));
typedef __bf16 v8bf  __attribute__((ext_vector_type(8)));
typedef float  v8f   __attribute__((ext_vector_type(8)));

union Frag { v16bf v; v8bf h[2]; };

__device__ __forceinline__ __bf16 f2bf(float f) {
  union { float f; unsigned u; } a; a.f = f;
  unsigned r = (a.u + 0x7FFFu + ((a.u >> 16) & 1u)) >> 16;   // round-to-nearest-even
  union { unsigned short s; __bf16 b; } o; o.s = (unsigned short)r;
  return o.b;
}

__device__ __forceinline__ float sigm(float x) {
  return 1.0f / (1.0f + __expf(-x));
}

// Async global->LDS copy of one 16-byte chunk per lane (CDNA5, ASYNCcnt-tracked).
// GVS mode: LDS address from vdst, global address is saddr + vaddr. One wave instruction moves 512B.
__device__ __forceinline__ void async_b128(unsigned lds_addr, unsigned voff, uint64_t sbase) {
  asm volatile("global_load_async_to_lds_b128 %0, %1, %2"
               :: "v"(lds_addr), "v"(voff), "s"(sbase) : "memory");
}
__device__ __forceinline__ void wait_async0() {
  asm volatile("s_wait_asynccnt 0x0" ::: "memory");
}

// ---------------------------------------------------------------------------
// Kernel 1: xW = X[M,K] @ W[K,N] + bias[N]   (M = B*T, N = 4H)
// Block: 256 threads (8 waves). Tile: 128(M) x 64(N). Full K depth in LDS.
// Wave w owns M-subtile w, iterates 4 N-subtiles; A/B staged as bf16 in LDS
// laid out so every WMMA fragment is two 16B contiguous LDS reads.
// ---------------------------------------------------------------------------
template <int K>
__global__ __launch_bounds__(256)
void lstm_gemm_xw(const float* __restrict__ A, const float* __restrict__ W,
                  const float* __restrict__ bias, float* __restrict__ out, int N) {
  extern __shared__ char smem[];
  __bf16* As = (__bf16*)smem;                         // [128][K] row-major
  __bf16* Bs = (__bf16*)(smem + 128 * K * 2);         // [64][K]  (W transposed)

  const int tid = threadIdx.x;
  const int m0  = blockIdx.x * 128;
  const int n0  = blockIdx.y * 64;

  // Stage A tile (f32 -> bf16)
  #pragma unroll 4
  for (int idx = tid; idx < 128 * K; idx += 256) {
    int r = idx / K, c = idx - r * K;                 // K is power of two
    As[idx] = f2bf(A[(size_t)(m0 + r) * K + c]);
  }
  // Stage W tile transposed: Bs[n][k] = W[k][n0+n]
  #pragma unroll 4
  for (int idx = tid; idx < 64 * K; idx += 256) {
    int n = idx / K, k = idx - n * K;
    Bs[idx] = f2bf(W[(size_t)k * N + n0 + n]);
  }
  __syncthreads();

  const int wave = tid >> 5, lane = tid & 31;
  const int mt   = wave;                 // M-subtile 0..7
  const int mrow = lane & 15;            // A-fragment row
  const int col  = lane & 15;            // B/C-fragment column
  const int koff = (lane >> 4) ? 8 : 0;  // K sub-offset per half-wave
  const int mtop = (lane >> 4) * 8;      // C-fragment row base

  v8f acc[4];
  #pragma unroll
  for (int nt = 0; nt < 4; ++nt) {
    float bv = bias[n0 + nt * 16 + col];
    #pragma unroll
    for (int v = 0; v < 8; ++v) acc[nt][v] = bv;
  }

  #pragma unroll
  for (int kt = 0; kt < K; kt += 32) {
    Frag a;
    const __bf16* ap = As + (mt * 16 + mrow) * K + kt + koff;
    a.h[0] = *(const v8bf*)(ap);
    a.h[1] = *(const v8bf*)(ap + 16);
    #pragma unroll
    for (int nt = 0; nt < 4; ++nt) {
      Frag b;
      const __bf16* bp = Bs + (nt * 16 + col) * K + kt + koff;
      b.h[0] = *(const v8bf*)(bp);
      b.h[1] = *(const v8bf*)(bp + 16);
      acc[nt] = __builtin_amdgcn_wmma_f32_16x16x32_bf16(
          false, a.v, false, b.v, (short)0, acc[nt], false, false);
    }
  }

  #pragma unroll
  for (int nt = 0; nt < 4; ++nt)
    #pragma unroll
    for (int v = 0; v < 8; ++v)
      out[(size_t)(m0 + mt * 16 + mtop + v) * N + n0 + nt * 16 + col] = acc[nt][v];
}

// ---------------------------------------------------------------------------
// Kernel 2: LSTM recurrence, persistent over all T steps.
// Grid: B/16 blocks; block = (H/16) waves (2H threads). Each block owns 16
// batch rows (recurrence is independent across batch). U^T lives in LDS as
// bf16 ([4H][H], up to 128 KB). The per-step xW tile [16][4H] f32 is
// double-buffered in LDS via GLOBAL_LOAD_ASYNC_TO_LDS_B128: step t+1's tile
// streams in (ASYNCcnt) while step t's WMMAs run; fenced with s_wait_asynccnt.
// Wave w owns a 16-column slice of each of the 4 gates, keeps cell state c
// in registers, shares h via a small LDS tile with two barriers per step.
// ---------------------------------------------------------------------------
template <int H>
__global__ __launch_bounds__((H / 16) * 32)
void lstm_rec(const float* __restrict__ xW, const float* __restrict__ U,
              float* __restrict__ hout, int T) {
  constexpr int G   = 4 * H;
  constexpr int NT  = (H / 16) * 32;     // threads per block
  constexpr int CPR = G / 4;             // 16B chunks per xW row (G*4/16)
  constexpr int NCH = 16 * CPR;          // chunks per tile
  constexpr int CPT = NCH / NT;          // chunks per thread (== 8)

  extern __shared__ char smem[];
  __bf16* Ut  = (__bf16*)smem;                              // [G][H]  U transposed
  __bf16* hb  = (__bf16*)(smem + (size_t)G * H * 2);        // [16][H] h_{t-1} (bf16)
  float*  xwf = (float*)(smem + (size_t)G * H * 2 + 16 * H * 2); // [2][16][G] xW tiles

  const int tid = threadIdx.x;
  const int b0  = blockIdx.x * 16;

  // Stage U transposed: Ut[n][k] = U[k][n]
  for (int idx = tid; idx < G * H; idx += NT) {
    int n = idx / H, k = idx - n * H;
    Ut[idx] = f2bf(U[(size_t)k * G + n]);
  }
  for (int idx = tid; idx < 16 * H; idx += NT) hb[idx] = f2bf(0.0f);

  // Per-thread chunk coordinates for the async xW tile copy
  const unsigned rowstride = (unsigned)(T * G * 4);
  const unsigned xw_lds0   = (unsigned)(uintptr_t)xwf;
  unsigned voffs[CPT], loffs[CPT];
  #pragma unroll
  for (int i = 0; i < CPT; ++i) {
    int c   = tid + i * NT;
    int row = c / CPR;                  // power-of-two divide
    int cw  = c - row * CPR;
    voffs[i] = (unsigned)row * rowstride + (unsigned)cw * 16u;
    loffs[i] = (unsigned)(row * G + cw * 4) * 4u;
  }

  // Prefetch t=0 tile into buffer 0
  {
    uint64_t base = (uint64_t)(uintptr_t)(xW + (size_t)b0 * T * G);
    #pragma unroll
    for (int i = 0; i < CPT; ++i) async_b128(xw_lds0 + loffs[i], voffs[i], base);
  }
  wait_async0();
  __syncthreads();

  const int wave = tid >> 5, lane = tid & 31;
  const int nc   = wave * 16;            // column slice within each gate
  const int col  = lane & 15;
  const int koff = (lane >> 4) ? 8 : 0;
  const int mrow = lane & 15;
  const int mtop = (lane >> 4) * 8;

  v8f c;
  #pragma unroll
  for (int v = 0; v < 8; ++v) c[v] = 0.0f;

  int buf = 0;
  for (int t = 0; t < T; ++t) {
    // Kick off async copy of next step's xW tile into the other buffer
    if (t + 1 < T) {
      uint64_t base = (uint64_t)(uintptr_t)(xW + ((size_t)b0 * T + (t + 1)) * G);
      unsigned dst  = xw_lds0 + (unsigned)((1 - buf) * 16 * G * 4);
      #pragma unroll
      for (int i = 0; i < CPT; ++i) async_b128(dst + loffs[i], voffs[i], base);
    }

    // C init = xW slice from LDS (input projection incl. bias)
    const float* xwc = xwf + (size_t)buf * 16 * G;
    v8f z[4];
    #pragma unroll
    for (int g = 0; g < 4; ++g) {
      const int gc = g * H + nc + col;
      #pragma unroll
      for (int v = 0; v < 8; ++v) z[g][v] = xwc[(mtop + v) * G + gc];
    }

    // z += h_{t-1} @ U
    #pragma unroll
    for (int kt = 0; kt < H; kt += 32) {
      Frag a;
      const __bf16* ap = hb + mrow * H + kt + koff;
      a.h[0] = *(const v8bf*)(ap);
      a.h[1] = *(const v8bf*)(ap + 16);
      #pragma unroll
      for (int g = 0; g < 4; ++g) {
        Frag b;
        const __bf16* bp = Ut + (size_t)(g * H + nc + col) * H + kt + koff;
        b.h[0] = *(const v8bf*)(bp);
        b.h[1] = *(const v8bf*)(bp + 16);
        z[g] = __builtin_amdgcn_wmma_f32_16x16x32_bf16(
            false, a.v, false, b.v, (short)0, z[g], false, false);
      }
    }
    __syncthreads();   // all waves done reading h_{t-1} from LDS

    // gates (i,f,g,o), relu cell activation per reference
    v8f hv;
    #pragma unroll
    for (int v = 0; v < 8; ++v) {
      float ig = sigm(z[0][v]);
      float fg = sigm(z[1][v]);
      float gg = fmaxf(z[2][v], 0.0f);
      float og = sigm(z[3][v]);
      float cn = fg * c[v] + ig * gg;
      c[v] = cn;
      hv[v] = og * fmaxf(cn, 0.0f);
    }
    #pragma unroll
    for (int v = 0; v < 8; ++v) {
      int m = mtop + v;
      hb[m * H + nc + col] = f2bf(hv[v]);
      hout[((size_t)(b0 + m) * T + t) * H + nc + col] = hv[v];
    }

    wait_async0();     // next tile fully landed in LDS (this wave's share)
    __syncthreads();   // h_t + all waves' async tile visible to everyone
    buf ^= 1;
  }
}

// ---------------------------------------------------------------------------
// Kernel 3: LayerNorm over the time axis (keras axis=1). One block per batch
// row, one thread per feature; t-loop is coalesced across the block.
// ---------------------------------------------------------------------------
__global__ void ln_time_k(float* __restrict__ x, const float* __restrict__ g,
                          const float* __restrict__ be, int H, int T) {
  const int b = blockIdx.x, f = threadIdx.x;
  const size_t base = (size_t)b * T * H + f;
  float s = 0.0f, ss = 0.0f;
  for (int t = 0; t < T; ++t) {
    float v = x[base + (size_t)t * H];
    s += v; ss += v * v;
  }
  const float inv_t = 1.0f / (float)T;
  const float mu  = s * inv_t;
  const float var = ss * inv_t - mu * mu;
  const float sc  = rsqrtf(var + 1e-3f);
  for (int t = 0; t < T; ++t) {
    size_t a = base + (size_t)t * H;
    x[a] = (x[a] - mu) * sc * g[t] + be[t];
  }
}

// ---------------------------------------------------------------------------
// Kernel 4: dense head. One thread per batch row; weights staged in LDS.
// h_last (64) -> relu(64) -> relu(32) -> softmax(3) ++ softmax(2)
// ---------------------------------------------------------------------------
__global__ __launch_bounds__(256)
void head_k(const float* __restrict__ hseq,
            const float* __restrict__ dw1, const float* __restrict__ db1,
            const float* __restrict__ dw2, const float* __restrict__ db2,
            const float* __restrict__ ow1, const float* __restrict__ ob1,
            const float* __restrict__ ow2, const float* __restrict__ ob2,
            float* __restrict__ out, int T) {
  __shared__ float W1[64 * 64], B1[64], W2[64 * 32], B2[32];
  __shared__ float O1[32 * 3], Q1[3], O2[32 * 2], Q2[2];
  const int tid = threadIdx.x;
  for (int i = tid; i < 64 * 64; i += 256) W1[i] = dw1[i];
  for (int i = tid; i < 64;      i += 256) B1[i] = db1[i];
  for (int i = tid; i < 64 * 32; i += 256) W2[i] = dw2[i];
  for (int i = tid; i < 32;      i += 256) B2[i] = db2[i];
  for (int i = tid; i < 32 * 3;  i += 256) O1[i] = ow1[i];
  for (int i = tid; i < 3;       i += 256) Q1[i] = ob1[i];
  for (int i = tid; i < 32 * 2;  i += 256) O2[i] = ow2[i];
  for (int i = tid; i < 2;       i += 256) Q2[i] = ob2[i];
  __syncthreads();

  const int b = blockIdx.x * 256 + tid;
  const float* h = hseq + ((size_t)b * T + (T - 1)) * 64;

  float h0[64];
  #pragma unroll
  for (int k = 0; k < 64; ++k) h0[k] = h[k];

  float h1[64];
  #pragma unroll 4
  for (int j = 0; j < 64; ++j) {
    float a = B1[j];
    #pragma unroll
    for (int k = 0; k < 64; ++k) a += h0[k] * W1[k * 64 + j];
    h1[j] = fmaxf(a, 0.0f);
  }
  float h2[32];
  #pragma unroll 4
  for (int j = 0; j < 32; ++j) {
    float a = B2[j];
    #pragma unroll
    for (int k = 0; k < 64; ++k) a += h1[k] * W2[k * 32 + j];
    h2[j] = fmaxf(a, 0.0f);
  }
  float o1[3];
  #pragma unroll
  for (int j = 0; j < 3; ++j) {
    float a = Q1[j];
    #pragma unroll
    for (int k = 0; k < 32; ++k) a += h2[k] * O1[k * 3 + j];
    o1[j] = a;
  }
  float o2[2];
  #pragma unroll
  for (int j = 0; j < 2; ++j) {
    float a = Q2[j];
    #pragma unroll
    for (int k = 0; k < 32; ++k) a += h2[k] * O2[k * 2 + j];
    o2[j] = a;
  }
  // softmax(3)
  float m1 = fmaxf(o1[0], fmaxf(o1[1], o1[2]));
  float e0 = __expf(o1[0] - m1), e1 = __expf(o1[1] - m1), e2 = __expf(o1[2] - m1);
  float r1 = 1.0f / (e0 + e1 + e2);
  // softmax(2)
  float m2 = fmaxf(o2[0], o2[1]);
  float f0 = __expf(o2[0] - m2), f1 = __expf(o2[1] - m2);
  float r2 = 1.0f / (f0 + f1);

  float* po = out + (size_t)b * 5;
  po[0] = e0 * r1; po[1] = e1 * r1; po[2] = e2 * r1;
  po[3] = f0 * r2; po[4] = f1 * r2;
}

// ---------------------------------------------------------------------------
// Host orchestration
// ---------------------------------------------------------------------------
extern "C" void kernel_launch(void* const* d_in, const int* in_sizes, int n_in,
                              void* d_out, int out_size, void* d_ws, size_t ws_size,
                              hipStream_t stream) {
  (void)in_sizes; (void)n_in; (void)out_size; (void)ws_size;
  const int B = 1024, T = 256;

  const float* x = (const float*)d_in[0];
  const float* Wl[6]; const float* Ul[6]; const float* bl[6];
  for (int i = 0; i < 6; ++i) {
    Wl[i] = (const float*)d_in[1 + 3 * i];
    Ul[i] = (const float*)d_in[2 + 3 * i];
    bl[i] = (const float*)d_in[3 + 3 * i];
  }
  const float* lg[3] = {(const float*)d_in[19], (const float*)d_in[21], (const float*)d_in[23]};
  const float* lb[3] = {(const float*)d_in[20], (const float*)d_in[22], (const float*)d_in[24]};
  const float* dw1 = (const float*)d_in[25]; const float* db1 = (const float*)d_in[26];
  const float* dw2 = (const float*)d_in[27]; const float* db2 = (const float*)d_in[28];
  const float* ow1 = (const float*)d_in[29]; const float* ob1 = (const float*)d_in[30];
  const float* ow2 = (const float*)d_in[31]; const float* ob2 = (const float*)d_in[32];
  float* out = (float*)d_out;

  // workspace: xW buffer [B,T,512] + two h-sequence ping-pong buffers [B,T,128]
  char* ws = (char*)d_ws;
  float* xw = (float*)ws;
  float* hA = (float*)(ws + (size_t)B * T * 512 * sizeof(float));
  float* hB = hA + (size_t)B * T * 128;

  const int din[6] = {32, 64, 128, 128, 64, 64};
  const int hh[6]  = {64, 128, 128, 64, 64, 64};

  const float* cur = x;
  float* curo = hA;
  int lnidx = 0;
  const int M = B * T;

  for (int l = 0; l < 6; ++l) {
    const int K = din[l], H = hh[l], N = 4 * H;

    dim3 gg(M / 128, N / 64);
    size_t smg = (size_t)(128 + 64) * K * 2;
    if (K == 32)
      lstm_gemm_xw<32><<<gg, 256, smg, stream>>>(cur, Wl[l], bl[l], xw, N);
    else if (K == 64)
      lstm_gemm_xw<64><<<gg, 256, smg, stream>>>(cur, Wl[l], bl[l], xw, N);
    else
      lstm_gemm_xw<128><<<gg, 256, smg, stream>>>(cur, Wl[l], bl[l], xw, N);

    // LDS: U^T (bf16) + h tile (bf16) + double-buffered xW tile (f32)
    size_t smr = (size_t)N * H * 2 + (size_t)16 * H * 2 + (size_t)2 * 16 * N * 4;
    if (H == 64)
      lstm_rec<64><<<B / 16, (64 / 16) * 32, smr, stream>>>(xw, Ul[l], curo, T);
    else
      lstm_rec<128><<<B / 16, (128 / 16) * 32, smr, stream>>>(xw, Ul[l], curo, T);

    if (l == 0 || l == 2 || l == 4) {
      ln_time_k<<<B, H, 0, stream>>>(curo, lg[lnidx], lb[lnidx], H, T);
      ++lnidx;
    }

    cur = curo;
    curo = (curo == hA) ? hB : hA;
  }

  head_k<<<B / 256, 256, 0, stream>>>(cur, dw1, db1, dw2, db2,
                                      ow1, ob1, ow2, ob2, out, T);
}
